// MedicalVisionTransformer_35742717837864
// MI455X (gfx1250) — compile-verified, matrix operand order checked by reading, not run on hardware
//
#include <hip/hip_runtime.h>
#include <math.h>

// ---------------- model dims ----------------
#define B_    64
#define SR_   29
#define S_    30
#define D_    768
#define H_    3072
#define L_    12
#define NH_   12
#define DH_   64      // D/NH
#define ND_   14
#define E_    15
#define RK_   8
#define NE_   6       // L/2
#define DK_   384     // D/2
#define LORA_SCALE 2.0f   // 16/8

typedef __bf16 bf16;
typedef bf16  v16bf __attribute__((ext_vector_type(16)));
typedef bf16  v4bf  __attribute__((ext_vector_type(4)));
typedef float v8f   __attribute__((ext_vector_type(8)));

__device__ __forceinline__ float gelu_exact(float x) {
    return 0.5f * x * (1.0f + erff(x * 0.70710678118654752440f));
}

__device__ __forceinline__ float block_reduce_sum(float v, float* red) {
    int tid = threadIdx.x;
    red[tid] = v;
    __syncthreads();
    for (int off = blockDim.x >> 1; off > 0; off >>= 1) {
        if (tid < off) red[tid] += red[tid + off];
        __syncthreads();
    }
    float r = red[0];
    __syncthreads();
    return r;
}

// ---------------- h = concat(cls, region_features) ----------------
__global__ __launch_bounds__(256)
void init_h_kernel(const float* __restrict__ rf, const float* __restrict__ cls,
                   float* __restrict__ h) {
    size_t n = (size_t)B_ * S_ * D_;
    for (size_t i = blockIdx.x * 256ull + threadIdx.x; i < n; i += (size_t)gridDim.x * 256ull) {
        int d = (int)(i % D_);
        size_t bs = i / D_;
        int s = (int)(bs % S_);
        int b = (int)(bs / S_);
        h[i] = (s == 0) ? cls[d] : rf[((size_t)b * SR_ + (s - 1)) * D_ + d];
    }
}

// ---------------- elementwise add ----------------
__global__ __launch_bounds__(256)
void add_kernel(const float* __restrict__ a, const float* __restrict__ b,
                float* __restrict__ y, size_t n) {
    for (size_t i = blockIdx.x * 256ull + threadIdx.x; i < n; i += (size_t)gridDim.x * 256ull)
        y[i] = a[i] + b[i];
}

// ---------------- LayerNorm (one block per row) ----------------
__global__ __launch_bounds__(256)
void layernorm_kernel(const float* __restrict__ x, const float* __restrict__ g,
                      const float* __restrict__ bta, float* __restrict__ y,
                      int d, float eps) {
    __shared__ float red[256];
    size_t row = blockIdx.x;
    const float* xr = x + row * d;
    float* yr = y + row * d;
    int tid = threadIdx.x;

    float s = 0.f;
    for (int i = tid; i < d; i += 256) s += xr[i];
    float mean = block_reduce_sum(s, red) / (float)d;

    float vs = 0.f;
    for (int i = tid; i < d; i += 256) { float t = xr[i] - mean; vs += t * t; }
    float var = block_reduce_sum(vs, red) / (float)d;
    float rstd = rsqrtf(var + eps);

    for (int i = tid; i < d; i += 256)
        yr[i] = (xr[i] - mean) * rstd * g[i] + bta[i];
}

// ---------------- WMMA bf16 GEMM ----------------
// C[z] = act(A @ W[z]^T + bias[z] [+ resid]); A:[M,K] f32, W:[N,K] f32.
// REQUIRES: M % 64 == 0, N % 256 == 0, K % 32 == 0 (true for all launches here).
// blockIdx.z in {0,1,2} selects (W,bias,C) -> lets q/k/v run as one launch.
#define BM 64
#define BN 256
#define BK 32
#define LDK 40

__global__ __launch_bounds__(256)
void gemm_wmma_bf16(const float* __restrict__ A,
                    const float* __restrict__ W0, const float* __restrict__ bv0, float* __restrict__ C0,
                    const float* __restrict__ W1, const float* __restrict__ bv1, float* __restrict__ C1,
                    const float* __restrict__ W2, const float* __restrict__ bv2, float* __restrict__ C2,
                    const float* __restrict__ resid, int M, int N, int K, int act) {
    __shared__ bf16 As[2][BM][LDK];
    __shared__ bf16 Bs[2][BN][LDK];

    const int z = blockIdx.z;
    const float* W    = (z == 0) ? W0 : (z == 1) ? W1 : W2;
    const float* bias = (z == 0) ? bv0 : (z == 1) ? bv1 : bv2;
    float*       C    = (z == 0) ? C0 : (z == 1) ? C1 : C2;

    const int tid   = threadIdx.x;
    const int lane  = tid & 31;
    const int wave  = tid >> 5;     // 0..7
    const int wr    = wave >> 2;    // 0..1 : 32-row band
    const int wc    = wave & 3;     // 0..3 : 64-col band
    const int group = lane >> 4;    // 0/1
    const int lrow  = lane & 15;

    const int bm0 = blockIdx.y * BM;
    const int bn0 = blockIdx.x * BN;

    v8f c[2][4];
#pragma unroll
    for (int i = 0; i < 2; ++i)
#pragma unroll
        for (int j = 0; j < 4; ++j)
#pragma unroll
            for (int r = 0; r < 8; ++r) c[i][j][r] = 0.0f;

    // staging geometry: float4 index f -> row = f>>3, col4 = (f&7)*4
    float4 ar[2], br[8];

    auto stage = [&](int k0) {
#pragma unroll
        for (int t = 0; t < 2; ++t) {
            int f = tid + 256 * t;
            ar[t] = *(const float4*)(A + (size_t)(bm0 + (f >> 3)) * K + k0 + (f & 7) * 4);
        }
#pragma unroll
        for (int t = 0; t < 8; ++t) {
            int f = tid + 256 * t;
            br[t] = *(const float4*)(W + (size_t)(bn0 + (f >> 3)) * K + k0 + (f & 7) * 4);
        }
    };
    auto commit = [&](int buf) {
#pragma unroll
        for (int t = 0; t < 2; ++t) {
            int f = tid + 256 * t;
            v4bf p;
            p[0] = (bf16)ar[t].x; p[1] = (bf16)ar[t].y;
            p[2] = (bf16)ar[t].z; p[3] = (bf16)ar[t].w;
            *(v4bf*)&As[buf][f >> 3][(f & 7) * 4] = p;
        }
#pragma unroll
        for (int t = 0; t < 8; ++t) {
            int f = tid + 256 * t;
            v4bf p;
            p[0] = (bf16)br[t].x; p[1] = (bf16)br[t].y;
            p[2] = (bf16)br[t].z; p[3] = (bf16)br[t].w;
            *(v4bf*)&Bs[buf][f >> 3][(f & 7) * 4] = p;
        }
    };

    stage(0);
    commit(0);
    int cur = 0;

    for (int k0 = 0; k0 < K; k0 += BK) {
        __syncthreads();
        const bool more = (k0 + BK) < K;
        if (more) stage(k0 + BK);                       // global loads in flight over WMMAs
        if (k0 + 2 * BK < K) {                          // L2 prefetch of tile after next
            __builtin_prefetch(A + (size_t)(bm0 + (tid >> 3)) * K + k0 + 2 * BK + (tid & 7) * 4, 0, 0);
            __builtin_prefetch(W + (size_t)(bn0 + (tid >> 3)) * K + k0 + 2 * BK + (tid & 7) * 4, 0, 0);
        }

        // Build fragments per CDNA5 16-bit layout:
        // element j -> K = 16*(j>=8) + ((j>>1)&3)*2 + (j&1) + 8*group
        v16bf afrag[2], bfrag[4];
#pragma unroll
        for (int t = 0; t < 2; ++t)
#pragma unroll
            for (int j = 0; j < 16; ++j) {
                int kk = ((j & 8) ? 16 : 0) + ((j >> 1) & 3) * 2 + (j & 1) + group * 8;
                afrag[t][j] = As[cur][wr * 32 + t * 16 + lrow][kk];
            }
#pragma unroll
        for (int u = 0; u < 4; ++u)
#pragma unroll
            for (int j = 0; j < 16; ++j) {
                int kk = ((j & 8) ? 16 : 0) + ((j >> 1) & 3) * 2 + (j & 1) + group * 8;
                bfrag[u][j] = Bs[cur][wc * 64 + u * 16 + lrow][kk];
            }

#pragma unroll
        for (int i = 0; i < 2; ++i)
#pragma unroll
            for (int u = 0; u < 4; ++u)
                c[i][u] = __builtin_amdgcn_wmma_f32_16x16x32_bf16(
                    false, afrag[i], false, bfrag[u], (short)0, c[i][u], false, false);

        if (more) commit(cur ^ 1);
        cur ^= 1;
    }

    // epilogue: VGPR r, lanes 0-15 -> M=r, lanes 16-31 -> M=r+8; N=lane&15
#pragma unroll
    for (int i = 0; i < 2; ++i)
#pragma unroll
        for (int u = 0; u < 4; ++u) {
            int n = bn0 + wc * 64 + u * 16 + lrow;
            float bb = bias ? bias[n] : 0.0f;
#pragma unroll
            for (int r = 0; r < 8; ++r) {
                int m = bm0 + wr * 32 + i * 16 + group * 8 + r;
                float v = c[i][u][r] + bb;
                if (resid) v += resid[(size_t)m * N + n];
                if (act)   v = gelu_exact(v);
                C[(size_t)m * N + n] = v;
            }
        }
}

// ---------------- attention: per (head, batch), S=30, dh=64 ----------------
__global__ __launch_bounds__(32)
void attention_kernel(const float* __restrict__ q, const float* __restrict__ k,
                      const float* __restrict__ v, float* __restrict__ ctx) {
    int hd = blockIdx.x;
    int b  = blockIdx.y;
    int lane = threadIdx.x;

    __shared__ float ks[S_][DH_];
    __shared__ float vs[S_][DH_];
    for (int i = lane; i < S_ * DH_; i += 32) {
        int s = i / DH_, d = i % DH_;
        size_t off = ((size_t)b * S_ + s) * D_ + hd * DH_ + d;
        ks[s][d] = k[off];
        vs[s][d] = v[off];
    }
    __syncthreads();

    if (lane < S_) {
        float qr[DH_];
        size_t qoff = ((size_t)b * S_ + lane) * D_ + hd * DH_;
#pragma unroll
        for (int d = 0; d < DH_; ++d) qr[d] = q[qoff + d];

        float sc[S_];
        float mx = -3.0e38f;
        for (int kk = 0; kk < S_; ++kk) {
            float dot = 0.f;
#pragma unroll
            for (int d = 0; d < DH_; ++d) dot += qr[d] * ks[kk][d];
            sc[kk] = dot * 0.125f;
            mx = fmaxf(mx, sc[kk]);
        }
        float o[DH_];
#pragma unroll
        for (int d = 0; d < DH_; ++d) o[d] = 0.f;
        float den = 0.f;
        for (int kk = 0; kk < S_; ++kk) {
            float w = expf(sc[kk] - mx);
            den += w;
#pragma unroll
            for (int d = 0; d < DH_; ++d) o[d] += w * vs[kk][d];
        }
        float inv = 1.0f / den;
#pragma unroll
        for (int d = 0; d < DH_; ++d) ctx[qoff + d] = o[d] * inv;
    }
}

// ---------------- region pooling ----------------
__global__ __launch_bounds__(256)
void pool_kernel(const float* __restrict__ mask, const float* __restrict__ attn,
                 float* __restrict__ pooled) {
    int d = blockIdx.x;
    int b = blockIdx.y;
    float cnt = 0.f;
    for (int r = 0; r < SR_; ++r) cnt += mask[r * ND_ + d];
    float inv = 1.0f / cnt;
    for (int hh = threadIdx.x; hh < D_; hh += 256) {
        float s = 0.f;
        for (int r = 0; r < SR_; ++r)
            s += mask[r * ND_ + d] * attn[((size_t)b * S_ + 1 + r) * D_ + hh];
        pooled[((size_t)b * ND_ + d) * D_ + hh] = s * inv;
    }
}

// ---------------- classifier head ----------------
__global__ __launch_bounds__(256)
void classifier_kernel(const float* __restrict__ pooled,
                       const float* __restrict__ cW1, const float* __restrict__ cb1,
                       const float* __restrict__ clng, const float* __restrict__ clnb,
                       const float* __restrict__ cW2, const float* __restrict__ cb2,
                       int el, float* __restrict__ dact) {
    __shared__ float z[DK_];
    __shared__ float red[256];
    int d = blockIdx.x;
    int b = blockIdx.y;
    int tid = threadIdx.x;

    const float* pr = pooled + ((size_t)b * ND_ + d) * D_;
    size_t base1 = ((size_t)el * ND_ + d) * DK_;

    for (int kk = tid; kk < DK_; kk += 256) {
        const float* wrow = cW1 + (base1 + kk) * D_;
        float s = 0.f;
        for (int hh = 0; hh < D_; ++hh) s += pr[hh] * wrow[hh];
        z[kk] = s + cb1[base1 + kk];
    }
    __syncthreads();

    float ps = 0.f, ps2 = 0.f;
    for (int kk = tid; kk < DK_; kk += 256) { float t = z[kk]; ps += t; ps2 += t * t; }
    float sum  = block_reduce_sum(ps, red);
    float sum2 = block_reduce_sum(ps2, red);
    float mean = sum / (float)DK_;
    float var  = sum2 / (float)DK_ - mean * mean;
    float rstd = rsqrtf(var + 1e-5f);

    for (int kk = tid; kk < DK_; kk += 256) {
        float t = (z[kk] - mean) * rstd * clng[base1 + kk] + clnb[base1 + kk];
        z[kk] = gelu_exact(t);
    }
    __syncthreads();

    float pp = 0.f;
    for (int kk = tid; kk < DK_; kk += 256) pp += z[kk] * cW2[base1 + kk];
    float pred = block_reduce_sum(pp, red);
    if (tid == 0) {
        pred += cb2[(size_t)el * ND_ + d];
        dact[(size_t)b * ND_ + d] = (pred > 0.0f) ? 1.0f : 0.0f;  // sigmoid(p)>0.5
    }
}

// ---------------- expert weights per (b,s) ----------------
__global__ __launch_bounds__(256)
void weights_kernel(const float* __restrict__ dact, const float* __restrict__ mask,
                    float* __restrict__ wts) {
    int n = B_ * S_;
    for (int i = blockIdx.x * 256 + threadIdx.x; i < n; i += gridDim.x * 256) {
        int s = i % S_;
        int b = i / S_;
        float act[E_];
        float sum = 0.f;
        for (int e = 0; e < E_; ++e) {
            float a;
            if (e == E_ - 1) a = 1.0f;
            else if (e < ND_ && s >= 1) a = dact[(size_t)b * ND_ + e] * mask[(s - 1) * ND_ + e];
            else a = 0.0f;
            act[e] = a;
            sum += a;
        }
        float inv = 1.0f / fmaxf(sum, 1.0f);
        for (int e = 0; e < E_; ++e) wts[(size_t)i * E_ + e] = act[e] * inv;
    }
}

// ---------------- fused LoRA mixture: one block per (b,s) ----------------
__global__ __launch_bounds__(256)
void lora_kernel(const float* __restrict__ n2, const float* __restrict__ inter,
                 const float* __restrict__ basep, const float* __restrict__ wts,
                 const float* __restrict__ A_up, const float* __restrict__ B_up,
                 const float* __restrict__ A_down, const float* __restrict__ B_down,
                 int el, float* __restrict__ ffn) {
    __shared__ float n2s[D_];
    __shared__ float inters[H_];
    __shared__ float bases[D_];
    __shared__ float part[RK_][257];
    __shared__ float red[256];

    size_t bs = blockIdx.x;
    int tid = threadIdx.x;

    for (int i = tid; i < D_; i += 256) {
        n2s[i]   = n2[bs * D_ + i];
        bases[i] = basep[bs * D_ + i];
    }
    for (int i = tid; i < H_; i += 256) inters[i] = inter[bs * H_ + i];
    __syncthreads();

    float accv[3] = {0.f, 0.f, 0.f};

    for (int e = 0; e < E_; ++e) {
        float w = wts[bs * E_ + e];
        if (w == 0.0f) continue;   // block-uniform branch
        size_t eb = (size_t)el * E_ + e;

        float p[RK_];
#pragma unroll
        for (int r = 0; r < RK_; ++r) p[r] = 0.f;
        for (int d = tid; d < D_; d += 256) {
            float x = n2s[d];
#pragma unroll
            for (int r = 0; r < RK_; ++r)
                p[r] += A_up[(eb * RK_ + r) * D_ + d] * x;
        }
#pragma unroll
        for (int r = 0; r < RK_; ++r) part[r][tid] = p[r];
        __syncthreads();
        for (int off = 128; off > 0; off >>= 1) {
            if (tid < off)
#pragma unroll
                for (int r = 0; r < RK_; ++r) part[r][tid] += part[r][tid + off];
            __syncthreads();
        }
        float rup[RK_];
#pragma unroll
        for (int r = 0; r < RK_; ++r) rup[r] = part[r][0];
        __syncthreads();

        float pd[RK_];
#pragma unroll
        for (int r = 0; r < RK_; ++r) pd[r] = 0.f;
        for (int hh = tid; hh < H_; hh += 256) {
            float lup = 0.f;
#pragma unroll
            for (int r = 0; r < RK_; ++r)
                lup += B_up[(eb * H_ + hh) * RK_ + r] * rup[r];
            float hid = inters[hh] + LORA_SCALE * lup;
#pragma unroll
            for (int r = 0; r < RK_; ++r)
                pd[r] += A_down[(eb * RK_ + r) * H_ + hh] * hid;
        }
#pragma unroll
        for (int r = 0; r < RK_; ++r) part[r][tid] = pd[r];
        __syncthreads();
        for (int off = 128; off > 0; off >>= 1) {
            if (tid < off)
#pragma unroll
                for (int r = 0; r < RK_; ++r) part[r][tid] += part[r][tid + off];
            __syncthreads();
        }
        float rdown[RK_];
#pragma unroll
        for (int r = 0; r < RK_; ++r) rdown[r] = part[r][0];
        __syncthreads();

        float ov[3];
        float s1 = 0.f, s2 = 0.f;
#pragma unroll
        for (int ci = 0; ci < 3; ++ci) {
            int d = tid + ci * 256;
            float ld = 0.f;
#pragma unroll
            for (int r = 0; r < RK_; ++r)
                ld += B_down[(eb * D_ + d) * RK_ + r] * rdown[r];
            float o = bases[d] + LORA_SCALE * ld;
            ov[ci] = o;
            s1 += o;
            s2 += o * o;
        }
        float sum  = block_reduce_sum(s1, red);
        float sum2 = block_reduce_sum(s2, red);
        float mean = sum / (float)D_;
        float var  = sum2 / (float)D_ - mean * mean;
        float rstd = rsqrtf(var + 1e-5f);
#pragma unroll
        for (int ci = 0; ci < 3; ++ci) accv[ci] += w * (ov[ci] - mean) * rstd;
    }

#pragma unroll
    for (int ci = 0; ci < 3; ++ci) ffn[bs * D_ + tid + ci * 256] = accv[ci];
}

// ---------------- host orchestration ----------------
extern "C" void kernel_launch(void* const* d_in, const int* in_sizes, int n_in,
                              void* d_out, int out_size, void* d_ws, size_t ws_size,
                              hipStream_t stream) {
    const float* rf    = (const float*)d_in[0];
    const float* mask  = (const float*)d_in[1];
    const float* cls   = (const float*)d_in[2];
    const float* ln1g  = (const float*)d_in[3];
    const float* ln1b  = (const float*)d_in[4];
    const float* ln2g  = (const float*)d_in[5];
    const float* ln2b  = (const float*)d_in[6];
    const float* Wq    = (const float*)d_in[7];
    const float* bq    = (const float*)d_in[8];
    const float* Wk    = (const float*)d_in[9];
    const float* bk    = (const float*)d_in[10];
    const float* Wv    = (const float*)d_in[11];
    const float* bv    = (const float*)d_in[12];
    const float* Wao   = (const float*)d_in[13];
    const float* bao   = (const float*)d_in[14];
    const float* Wi    = (const float*)d_in[15];
    const float* bi    = (const float*)d_in[16];
    const float* Wo    = (const float*)d_in[17];
    const float* bo    = (const float*)d_in[18];
    const float* fg    = (const float*)d_in[19];
    const float* fb    = (const float*)d_in[20];
    const float* cW1   = (const float*)d_in[21];
    const float* cb1   = (const float*)d_in[22];
    const float* clng  = (const float*)d_in[23];
    const float* clnb  = (const float*)d_in[24];
    const float* cW2   = (const float*)d_in[25];
    const float* cb2   = (const float*)d_in[26];
    const float* A_up  = (const float*)d_in[27];
    const float* B_up  = (const float*)d_in[28];
    const float* A_dn  = (const float*)d_in[29];
    const float* B_dn  = (const float*)d_in[30];
    (void)in_sizes; (void)n_in; (void)out_size; (void)ws_size;

    const size_t BSD = (size_t)B_ * S_ * D_;
    const size_t BSH = (size_t)B_ * S_ * H_;
    float* ws = (float*)d_ws;
    float* h      = ws;            ws += BSD;
    float* n1     = ws;            ws += BSD;
    float* q      = ws;            ws += BSD;
    float* k      = ws;            ws += BSD;
    float* v      = ws;            ws += BSD;
    float* ctx    = ws;            ws += BSD;
    float* attn   = ws;            ws += BSD;
    float* res1   = ws;            ws += BSD;
    float* n2     = ws;            ws += BSD;
    float* basep  = ws;            ws += BSD;
    float* ffn    = ws;            ws += BSD;
    float* inter  = ws;            ws += BSH;
    float* pooled = ws;            ws += (size_t)B_ * ND_ * D_;
    float* dact   = ws;            ws += (size_t)B_ * ND_;
    float* wts    = ws;            ws += (size_t)B_ * S_ * E_;

    const int M = B_ * S_;                 // 1920
    dim3 gQKV(D_ / BN, M / BM, 3);         // (3, 30, 3)
    dim3 gProjD(D_ / BN, M / BM, 1);       // (3, 30)
    dim3 gProjH(H_ / BN, M / BM, 1);       // (12, 30)
    int nBlkE = (int)((BSD + 255) / 256);

    init_h_kernel<<<512, 256, 0, stream>>>(rf, cls, h);

    for (int i = 0; i < L_; ++i) {
        const size_t wD = (size_t)i * D_ * D_;
        const size_t wH = (size_t)i * H_ * D_;
        const size_t iD = (size_t)i * D_;
        const size_t iH = (size_t)i * H_;
        const bool even = (i & 1) == 0;

        layernorm_kernel<<<M, 256, 0, stream>>>(h, ln1g + iD, ln1b + iD, n1, D_, 1e-12f);

        // fused q/k/v projection (blockIdx.z selects weight set)
        gemm_wmma_bf16<<<gQKV, 256, 0, stream>>>(n1,
            Wq + wD, bq + iD, q,
            Wk + wD, bk + iD, k,
            Wv + wD, bv + iD, v,
            nullptr, M, D_, D_, 0);

        attention_kernel<<<dim3(NH_, B_), 32, 0, stream>>>(q, k, v, ctx);

        if (even) {
            // need pre-residual attn for pooling
            gemm_wmma_bf16<<<gProjD, 256, 0, stream>>>(ctx,
                Wao + wD, bao + iD, attn,
                Wao + wD, bao + iD, attn,
                Wao + wD, bao + iD, attn,
                nullptr, M, D_, D_, 0);
            add_kernel<<<nBlkE, 256, 0, stream>>>(attn, h, res1, BSD);
        } else {
            // fuse residual: res1 = ctx@Wao^T + bao + h
            gemm_wmma_bf16<<<gProjD, 256, 0, stream>>>(ctx,
                Wao + wD, bao + iD, res1,
                Wao + wD, bao + iD, res1,
                Wao + wD, bao + iD, res1,
                h, M, D_, D_, 0);
        }

        layernorm_kernel<<<M, 256, 0, stream>>>(res1, ln2g + iD, ln2b + iD, n2, D_, 1e-12f);

        gemm_wmma_bf16<<<gProjH, 256, 0, stream>>>(n2,
            Wi + wH, bi + iH, inter,
            Wi + wH, bi + iH, inter,
            Wi + wH, bi + iH, inter,
            nullptr, M, H_, D_, 1);

        if (even) {
            gemm_wmma_bf16<<<gProjD, 256, 0, stream>>>(inter,
                Wo + wH, bo + iD, basep,
                Wo + wH, bo + iD, basep,
                Wo + wH, bo + iD, basep,
                nullptr, M, D_, H_, 0);

            int el = i >> 1;
            pool_kernel<<<dim3(ND_, B_), 256, 0, stream>>>(mask, attn, pooled);
            classifier_kernel<<<dim3(ND_, B_), 256, 0, stream>>>(pooled, cW1, cb1, clng, clnb, cW2, cb2, el, dact);
            weights_kernel<<<8, 256, 0, stream>>>(dact, mask, wts);
            lora_kernel<<<M, 256, 0, stream>>>(n2, inter, basep, wts,
                                               A_up, B_up, A_dn, B_dn, el, ffn);
            add_kernel<<<nBlkE, 256, 0, stream>>>(ffn, res1, h, BSD);
        } else {
            // fuse residual: h = inter@Wo^T + bo + res1
            gemm_wmma_bf16<<<gProjD, 256, 0, stream>>>(inter,
                Wo + wH, bo + iD, h,
                Wo + wH, bo + iD, h,
                Wo + wH, bo + iD, h,
                res1, M, D_, H_, 0);
        }
    }

    layernorm_kernel<<<M, 256, 0, stream>>>(h, fg, fb, (float*)d_out, D_, 1e-12f);
}
